// VelocityModel_88957362635102
// MI455X (gfx1250) — matrix operand order repaired and checked by reading, not compiled, not run on hardware
//
#include <hip/hip_runtime.h>
#include <hip/hip_bf16.h>
#include <math.h>

// ---------------------------------------------------------------------------
// CNF forward (CDNA5 / gfx1250, wave32, bf16 WMMA 16x16x32, f32 accumulate)
// Wave tile 32x64 (2x4 WMMA tiles), block = 8 waves (4M x 2N) = 128x128 tile.
// ---------------------------------------------------------------------------
#define B_SZ 2048
#define D_SZ 784
#define H_SZ 1024
#define T_SZ 50
#define KP   800    // D padded to a multiple of 32 for the K dim of GEMM #1
#define NPDZ 896    // dz-GEMM N (=784) padded to a multiple of 128

typedef __bf16 bf16_t;
typedef __bf16 bf16x8  __attribute__((ext_vector_type(8)));
typedef __bf16 bf16x16 __attribute__((ext_vector_type(16)));
typedef float  f32x8   __attribute__((ext_vector_type(8)));

__device__ __forceinline__ void atomAddF(float* p, float v) {
  __hip_atomic_fetch_add(p, v, __ATOMIC_RELAXED, __HIP_MEMORY_SCOPE_AGENT);
}

struct EpiArgs {
  const float*  bias;    // TANH: b ; DZ: b3
  const float*  tvec;    // TANH(HAS_T): W1t
  float         tval;    // TANH(HAS_T): t
  bf16_t*       outH;    // TANH: tanh() result (bf16)
  bf16_t*       outD;    // TANH: 1 - h^2 (bf16)
  const bf16_t* d2;      // TRACE: d2 (B x H, bf16)
  float*        logp;    // TRACE: accumulate -dt * trace
  float         dt;      // TRACE / DZ
  const float*  zprev;   // DZ: z_traj slot i   (B x 784, f32)
  float*        znext;   // DZ: z_traj slot i+1 (B x 784, f32)
  bf16_t*       zbf;     // DZ: padded bf16 z for next step (B x KP)
  int           nreal;   // DZ: real N (784)
};

enum { EPI_TANH = 0, EPI_TRACE = 1, EPI_DZ = 2 };

#define WM 2   // 16-row tiles per wave (M)
#define WN 4   // 16-col tiles per wave (N)

// D = X(MxK) * W(NxK)^T : both row-major, contiguous in K ("NT" GEMM).
// Grid: (N/128, M/128). Block 256 threads = 8 waves arranged 4(M) x 2(N).
template <int EPI, bool HAS_T>
__launch_bounds__(256)
__global__ void gemm_bf16_wmma(const bf16_t* __restrict__ X, int ldx,
                               const bf16_t* __restrict__ W, int ldw,
                               int K, EpiArgs ea) {
  const int lane = threadIdx.x & 31;
  const int wid  = threadIdx.x >> 5;
  const int wm   = wid >> 1;        // 0..3
  const int wn   = wid & 1;         // 0..1
  const int half = lane >> 4;       // 0/1
  const int l16  = lane & 15;

  const int mBase = blockIdx.y * 128 + wm * (WM * 16);
  const int nBase = blockIdx.x * 128 + wn * (WN * 16);

  f32x8 acc[WM][WN];
#pragma unroll
  for (int i = 0; i < WM; ++i)
#pragma unroll
    for (int j = 0; j < WN; ++j)
      acc[i][j] = (f32x8){0.f,0.f,0.f,0.f,0.f,0.f,0.f,0.f};

  for (int k = 0; k < K; k += 32) {
    bf16x16 a[WM], b[WN];
    // A fragment: lane holds row (mBase+i*16+l16), K chunks
    //   [k+8*half .. +7] and [k+16+8*half .. +7]  (ISA 16-bit A layout)
#pragma unroll
    for (int i = 0; i < WM; ++i) {
      const bf16_t* p = X + (size_t)(mBase + i * 16 + l16) * ldx + k + half * 8;
      bf16x8 lo = *(const bf16x8*)(p);
      bf16x8 hi = *(const bf16x8*)(p + 16);
#pragma unroll
      for (int e = 0; e < 8; ++e) { a[i][e] = lo[e]; a[i][e + 8] = hi[e]; }
    }
    // B fragment: lane holds output column n = nBase+j*16+l16, i.e. row n of
    // W, K chunks [k+8*half .. +7] and [k+16+8*half .. +7]  (same as A)
#pragma unroll
    for (int j = 0; j < WN; ++j) {
      const bf16_t* p = W + (size_t)(nBase + j * 16 + l16) * ldw + k + half * 8;
      bf16x8 lo = *(const bf16x8*)(p);
      bf16x8 hi = *(const bf16x8*)(p + 16);
#pragma unroll
      for (int e = 0; e < 8; ++e) { b[j][e] = lo[e]; b[j][e + 8] = hi[e]; }
    }
#pragma unroll
    for (int i = 0; i < WM; ++i)
#pragma unroll
      for (int j = 0; j < WN; ++j)
        acc[i][j] = __builtin_amdgcn_wmma_f32_16x16x32_bf16(
            false, a[i], false, b[j], (short)0, acc[i][j], false, false);
  }

  // C/D layout: VGPR r, lane -> row (r + 8*half), col l16.
  if (EPI == EPI_TANH) {
#pragma unroll
    for (int j = 0; j < WN; ++j) {
      const int col = nBase + j * 16 + l16;
      float bb = ea.bias[col];
      if (HAS_T) bb += ea.tval * ea.tvec[col];
#pragma unroll
      for (int i = 0; i < WM; ++i)
#pragma unroll
        for (int r = 0; r < 8; ++r) {
          const int row = mBase + i * 16 + r + half * 8;
          const float h = tanhf(acc[i][j][r] + bb);
          ea.outH[(size_t)row * H_SZ + col] = (bf16_t)h;
          ea.outD[(size_t)row * H_SZ + col] = (bf16_t)(1.f - h * h);
        }
    }
  } else if (EPI == EPI_TRACE) {
#pragma unroll
    for (int i = 0; i < WM; ++i)
#pragma unroll
      for (int r = 0; r < 8; ++r) {
        const int row = mBase + i * 16 + r + half * 8;
        float pr = 0.f;
#pragma unroll
        for (int j = 0; j < WN; ++j) {
          const int col = nBase + j * 16 + l16;
          pr += acc[i][j][r] * (float)ea.d2[(size_t)row * H_SZ + col];
        }
        pr += __shfl_xor(pr, 1, 32);
        pr += __shfl_xor(pr, 2, 32);
        pr += __shfl_xor(pr, 4, 32);
        pr += __shfl_xor(pr, 8, 32);
        if (l16 == 0) atomAddF(&ea.logp[row], -ea.dt * pr);  // logp += dt*(-tr)
      }
  } else {  // EPI_DZ
#pragma unroll
    for (int j = 0; j < WN; ++j) {
      const int col = nBase + j * 16 + l16;
      if (col < ea.nreal) {
        const float b3v = ea.bias[col];
#pragma unroll
        for (int i = 0; i < WM; ++i)
#pragma unroll
          for (int r = 0; r < 8; ++r) {
            const int row = mBase + i * 16 + r + half * 8;
            const float dz = acc[i][j][r] + b3v;
            const float zn = ea.zprev[(size_t)row * ea.nreal + col] + ea.dt * dz;
            ea.znext[(size_t)row * ea.nreal + col] = zn;
            ea.zbf[(size_t)row * KP + col] = (bf16_t)zn;
          }
      }
    }
  }
}

// ----------------------- setup / conversion kernels ------------------------
__global__ void k_conv_z(const float* __restrict__ z, bf16_t* __restrict__ zbf) {
  int idx = blockIdx.x * 256 + threadIdx.x;           // B_SZ*KP threads
  int c = idx % KP, b = idx / KP;
  zbf[idx] = (c < D_SZ) ? (bf16_t)z[(size_t)b * D_SZ + c] : (bf16_t)0.f;
}
__global__ void k_conv_w1(const float* __restrict__ W1, bf16_t* __restrict__ w1bf,
                          float* __restrict__ w1t) {
  int idx = blockIdx.x * 256 + threadIdx.x;           // H_SZ*KP threads
  int c = idx % KP, n = idx / KP;
  w1bf[idx] = (c < D_SZ) ? (bf16_t)W1[(size_t)n * (D_SZ + 1) + c] : (bf16_t)0.f;
  if (c == 0) w1t[n] = W1[(size_t)n * (D_SZ + 1) + D_SZ];
}
__global__ void k_conv_w2(const float* __restrict__ W2, bf16_t* __restrict__ o) {
  int idx = blockIdx.x * 256 + threadIdx.x;           // H_SZ*H_SZ threads
  o[idx] = (bf16_t)W2[idx];
}
__global__ void k_conv_w3(const float* __restrict__ W3, bf16_t* __restrict__ o) {
  int idx = blockIdx.x * 256 + threadIdx.x;           // NPDZ*H_SZ threads
  int n = idx / H_SZ;                                  // row (output dim), padded
  o[idx] = (n < D_SZ) ? (bf16_t)W3[idx] : (bf16_t)0.f; // W3 row-major (784,1024)
}
// A[j,k] = W2[j,k] * sum_d W1z[k,d] * W3[d,j]   (step-invariant)
__global__ void k_precompute_A(const float* __restrict__ W1, const float* __restrict__ W3,
                               const float* __restrict__ W2, bf16_t* __restrict__ Abf) {
  const int j = blockIdx.x * 256 + threadIdx.x;   // coalesced over W3 columns
  const int k = blockIdx.y;
  float s = 0.f;
  for (int d = 0; d < D_SZ; ++d)
    s = fmaf(W1[(size_t)k * (D_SZ + 1) + d], W3[(size_t)d * H_SZ + j], s);
  Abf[(size_t)j * H_SZ + k] = (bf16_t)(W2[(size_t)j * H_SZ + k] * s);
}
__global__ void k_zero(float* __restrict__ p) {
  p[blockIdx.x * 256 + threadIdx.x] = 0.f;
}
__global__ void k_logpx(const float* __restrict__ z0, const float* __restrict__ logp,
                        float* __restrict__ out) {
  __shared__ float sm[256];
  const int b = blockIdx.x;
  float s = 0.f;
  for (int d = threadIdx.x; d < D_SZ; d += 256) {
    float v = z0[(size_t)b * D_SZ + d];
    s += v * v;
  }
  sm[threadIdx.x] = s;
  __syncthreads();
  for (int st = 128; st > 0; st >>= 1) {
    if (threadIdx.x < st) sm[threadIdx.x] += sm[threadIdx.x + st];
    __syncthreads();
  }
  if (threadIdx.x == 0) {
    const float LOG2PI = 1.8378770664093453f;
    // NB: reference uses z_traj.shape[1] == B (2048) here — replicate exactly.
    out[b] = -0.5f * sm[0] - 0.5f * LOG2PI * (float)B_SZ - logp[b];
  }
}

// --------------------------------- launch ----------------------------------
extern "C" void kernel_launch(void* const* d_in, const int* in_sizes, int n_in,
                              void* d_out, int out_size, void* d_ws, size_t ws_size,
                              hipStream_t stream) {
  const float* z  = (const float*)d_in[0];
  const float* W1 = (const float*)d_in[1];
  const float* b1 = (const float*)d_in[2];
  const float* W2 = (const float*)d_in[3];
  const float* b2 = (const float*)d_in[4];
  const float* W3 = (const float*)d_in[5];
  const float* b3 = (const float*)d_in[6];
  float* out = (float*)d_out;

  // workspace carve-up (256B aligned)
  char* wsp = (char*)d_ws;
  size_t off = 0;
  auto carve = [&](size_t bytes) { void* p = wsp + off; off = (off + bytes + 255) & ~(size_t)255; return p; };
  bf16_t* zbf  = (bf16_t*)carve((size_t)B_SZ * KP * 2);
  bf16_t* w1bf = (bf16_t*)carve((size_t)H_SZ * KP * 2);
  float*  w1t  = (float*) carve((size_t)H_SZ * 4);
  bf16_t* w2bf = (bf16_t*)carve((size_t)H_SZ * H_SZ * 2);
  bf16_t* w3bf = (bf16_t*)carve((size_t)NPDZ * H_SZ * 2);
  bf16_t* abf  = (bf16_t*)carve((size_t)H_SZ * H_SZ * 2);
  bf16_t* h1bf = (bf16_t*)carve((size_t)B_SZ * H_SZ * 2);
  bf16_t* d1bf = (bf16_t*)carve((size_t)B_SZ * H_SZ * 2);
  bf16_t* h2bf = (bf16_t*)carve((size_t)B_SZ * H_SZ * 2);
  bf16_t* d2bf = (bf16_t*)carve((size_t)B_SZ * H_SZ * 2);
  float*  logp = (float*) carve((size_t)B_SZ * 4);

  // ---- one-time (per launch) setup ----
  k_conv_z <<<(B_SZ * KP) / 256, 256, 0, stream>>>(z, zbf);
  k_conv_w1<<<(H_SZ * KP) / 256, 256, 0, stream>>>(W1, w1bf, w1t);
  k_conv_w2<<<(H_SZ * H_SZ) / 256, 256, 0, stream>>>(W2, w2bf);
  k_conv_w3<<<(NPDZ * H_SZ) / 256, 256, 0, stream>>>(W3, w3bf);
  k_precompute_A<<<dim3(H_SZ / 256, H_SZ), 256, 0, stream>>>(W1, W3, W2, abf);
  k_zero<<<B_SZ / 256, 256, 0, stream>>>(logp);
  // z_traj[0] = z
  hipMemcpyAsync(out, z, (size_t)B_SZ * D_SZ * 4, hipMemcpyDeviceToDevice, stream);

  // ---- 49 Euler steps ----
  const float dt = 1.f / 49.f;
  const dim3 gH(H_SZ / 128, B_SZ / 128);   // N=1024 GEMMs
  const dim3 gZ(NPDZ / 128, B_SZ / 128);   // dz GEMM (N padded to 896)
  for (int i = 0; i < 49; ++i) {
    const float t = (float)i / 49.f;
    EpiArgs ea;

    // h1 = tanh(z @ W1z^T + t*W1t + b1); d1 = 1-h1^2
    ea = {};
    ea.bias = b1; ea.tvec = w1t; ea.tval = t; ea.outH = h1bf; ea.outD = d1bf;
    gemm_bf16_wmma<EPI_TANH, true><<<gH, 256, 0, stream>>>(zbf, KP, w1bf, KP, KP, ea);

    // h2 = tanh(h1 @ W2^T + b2); d2 = 1-h2^2
    ea = {};
    ea.bias = b2; ea.outH = h2bf; ea.outD = d2bf;
    gemm_bf16_wmma<EPI_TANH, false><<<gH, 256, 0, stream>>>(h1bf, H_SZ, w2bf, H_SZ, H_SZ, ea);

    // trace: v = d1 @ A^T ; logp += dt * -(sum_j d2*v)
    ea = {};
    ea.d2 = d2bf; ea.logp = logp; ea.dt = dt;
    gemm_bf16_wmma<EPI_TRACE, false><<<gH, 256, 0, stream>>>(d1bf, H_SZ, abf, H_SZ, H_SZ, ea);

    // dz = h2 @ W3^T + b3 ; z_{i+1} = z_i + dt*dz (f32 traj + bf16 staging)
    ea = {};
    ea.bias = b3; ea.dt = dt; ea.nreal = D_SZ;
    ea.zprev = out + (size_t)i * B_SZ * D_SZ;
    ea.znext = out + (size_t)(i + 1) * B_SZ * D_SZ;
    ea.zbf = zbf;
    gemm_bf16_wmma<EPI_DZ, false><<<gZ, 256, 0, stream>>>(h2bf, H_SZ, w3bf, H_SZ, H_SZ, ea);
  }

  // log_px = -0.5*||z0||^2 - 0.5*log(2pi)*B - logp
  k_logpx<<<B_SZ, 256, 0, stream>>>(out + (size_t)49 * B_SZ * D_SZ, logp,
                                    out + (size_t)T_SZ * B_SZ * D_SZ);
}
